// AttnBlock_89378269429864
// MI455X (gfx1250) — matrix operand order repaired and checked by reading, not compiled, not run on hardware
//
#include <hip/hip_runtime.h>
#include <hip/hip_bf16.h>
#include <stdint.h>

// ---------------------------------------------------------------------------
// Types for WMMA fragments (wave32, CDNA5 / gfx1250)
// ---------------------------------------------------------------------------
typedef _Float16 v16h __attribute__((ext_vector_type(16)));
typedef _Float16 v2h  __attribute__((ext_vector_type(2)));
typedef float    v8f  __attribute__((ext_vector_type(8)));

__device__ __forceinline__ v8f wmma_f16(v16h a, v16h b, v8f c) {
  // v_wmma_f32_16x16x32_f16  (neg_a, A, neg_b, B, c_mod, C, reuse_a, reuse_b)
  return __builtin_amdgcn_wmma_f32_16x16x32_f16(false, a, false, b, (short)0, c,
                                                false, false);
}

// Async global->LDS copy (CDNA5): 16 bytes, tracked by ASYNCcnt.
// ldsOff = DS byte address (low 32 bits of generic pointer to __shared__).
__device__ __forceinline__ void async_copy_b128(uint32_t ldsOff,
                                                const void* gaddr) {
  asm volatile("global_load_async_to_lds_b128 %0, %1, off"
               :
               : "v"(ldsOff), "v"((uint64_t)(uintptr_t)gaddr)
               : "memory");
}
__device__ __forceinline__ void wait_async0() {
  asm volatile("s_wait_asynccnt 0x0" ::: "memory");
}
__device__ __forceinline__ uint32_t lds_addr(const void* p) {
  return (uint32_t)(uintptr_t)p;  // LDS aperture low 32 bits == DS offset
}

// Problem constants
#define BATCH 4
#define CCH   256     // channels c
#define NTOK  2304    // 48*48
#define BN    (BATCH * NTOK)   // 9216 rows
#define INNER 512     // heads*dim_head = 2*c
#define NHEAD 8
#define DHEAD 64

// ---------------------------------------------------------------------------
// f32 -> f16 conversion (weights)
// ---------------------------------------------------------------------------
__global__ void f32_to_f16_kernel(const float* __restrict__ src,
                                  _Float16* __restrict__ dst, int n) {
  int i = blockIdx.x * 256 + threadIdx.x;
  if (i < n) dst[i] = (_Float16)src[i];
}

// ---------------------------------------------------------------------------
// Transpose x[b,c,h,w] -> T[b*n, c] (f32 residual) and LayerNorm -> TN (f16)
// ---------------------------------------------------------------------------
__global__ __launch_bounds__(256) void transpose_ln_kernel(
    const float* __restrict__ x, const float* __restrict__ g,
    const float* __restrict__ b, float* __restrict__ T,
    _Float16* __restrict__ TN) {
  const int row = blockIdx.x;            // b*NTOK + p
  const int bb  = row / NTOK;
  const int p   = row - bb * NTOK;
  const int ch  = threadIdx.x;
  float v = x[((size_t)bb * CCH + ch) * NTOK + p];
  T[(size_t)row * CCH + ch] = v;

  __shared__ float s1[256];
  __shared__ float s2[256];
  s1[ch] = v; s2[ch] = v * v;
  __syncthreads();
  for (int st = 128; st > 0; st >>= 1) {
    if (ch < st) { s1[ch] += s1[ch + st]; s2[ch] += s2[ch + st]; }
    __syncthreads();
  }
  float mean = s1[0] * (1.0f / CCH);
  float var  = s2[0] * (1.0f / CCH) - mean * mean;
  float rs   = rsqrtf(var + 1e-5f);
  TN[(size_t)row * CCH + ch] = (_Float16)((v - mean) * rs * g[ch] + b[ch]);
}

// LayerNorm of contiguous f32 rows -> f16
__global__ __launch_bounds__(256) void ln_kernel(
    const float* __restrict__ in, const float* __restrict__ g,
    const float* __restrict__ b, _Float16* __restrict__ out) {
  const int row = blockIdx.x;
  const int ch  = threadIdx.x;
  float v = in[(size_t)row * CCH + ch];
  __shared__ float s1[256];
  __shared__ float s2[256];
  s1[ch] = v; s2[ch] = v * v;
  __syncthreads();
  for (int st = 128; st > 0; st >>= 1) {
    if (ch < st) { s1[ch] += s1[ch + st]; s2[ch] += s2[ch + st]; }
    __syncthreads();
  }
  float mean = s1[0] * (1.0f / CCH);
  float var  = s2[0] * (1.0f / CCH) - mean * mean;
  float rs   = rsqrtf(var + 1e-5f);
  out[(size_t)row * CCH + ch] = (_Float16)((v - mean) * rs * g[ch] + b[ch]);
}

// ---------------------------------------------------------------------------
// WMMA GEMM: C[M,N] = A[M,K](f16) * B[K,N](f16)  + epilogue
// Block tile 64x128, 8 waves of 32x32, K-step 32, async-LDS staged operands.
// Epilogues: 0: store f16
//            1: +bias +res(f32) -> f32
//            2: +bias, exact GELU -> f16
//            3: +bias +res(f32) -> f16
//            4: conv-out: +bias[col] + x[b,col,p], transposed store -> f32
// ---------------------------------------------------------------------------
template <int EP>
__global__ __launch_bounds__(256) void gemm_kernel(
    const _Float16* __restrict__ A, const _Float16* __restrict__ Bw,
    const float* __restrict__ bias, const float* __restrict__ res,
    float* __restrict__ outF, _Float16* __restrict__ outH,
    const float* __restrict__ xin, int M, int N, int K) {
  __shared__ _Float16 As[64][48];    // row 96B (16B multiple)
  __shared__ _Float16 Bs[32][136];   // row 272B (16B multiple)

  const int tid  = threadIdx.x;
  const int lane = tid & 31;
  const int wave = tid >> 5;
  const int wy   = wave >> 2;   // 0..1
  const int wx   = wave & 3;    // 0..3
  const int m0   = blockIdx.y * 64;
  const int n0   = blockIdx.x * 128;
  const int hlf  = lane >> 4;   // 0/1
  const int lm   = lane & 15;

  // per-thread staging coordinates
  const int ar  = tid >> 2;          // A row 0..63
  const int acg = (tid & 3) * 8;     // A col group (halfs)
  const int br  = tid >> 3;          // B row 0..31
  const int bcg = (tid & 7) * 16;    // B col group (halfs)
  const uint32_t aDst  = lds_addr(&As[ar][acg]);
  const uint32_t bDst0 = lds_addr(&Bs[br][bcg]);
  const uint32_t bDst1 = lds_addr(&Bs[br][bcg + 8]);

  v8f acc[2][2];
  acc[0][0] = {}; acc[0][1] = {}; acc[1][0] = {}; acc[1][1] = {};

  for (int k0 = 0; k0 < K; k0 += 32) {
    // stage A (64x32) and B (32x128) via async global->LDS b128
    async_copy_b128(aDst, A + (size_t)(m0 + ar) * K + k0 + acg);
    const _Float16* bsrc = Bw + (size_t)(k0 + br) * N + n0 + bcg;
    async_copy_b128(bDst0, bsrc);
    async_copy_b128(bDst1, bsrc + 8);
    wait_async0();
    __syncthreads();

    v16h aF[2], bF[2];
#pragma unroll
    for (int i = 0; i < 2; ++i) {
      int m = wy * 32 + i * 16 + lm;
#pragma unroll
      for (int v = 0; v < 8; ++v) {
        int kk = (v < 4) ? (hlf * 8 + 2 * v) : (16 + hlf * 8 + 2 * (v - 4));
        v2h pr = *(const v2h*)&As[m][kk];
        aF[i][2 * v]     = pr[0];
        aF[i][2 * v + 1] = pr[1];
      }
    }
#pragma unroll
    for (int j = 0; j < 2; ++j) {
      int n = wx * 32 + j * 16 + lm;
#pragma unroll
      for (int v = 0; v < 8; ++v) {
        int kb = (v < 4) ? (hlf * 8 + 2 * v) : (16 + hlf * 8 + 2 * (v - 4));
        bF[j][2 * v]     = Bs[kb][n];
        bF[j][2 * v + 1] = Bs[kb + 1][n];
      }
    }
#pragma unroll
    for (int i = 0; i < 2; ++i)
#pragma unroll
      for (int j = 0; j < 2; ++j)
        acc[i][j] = wmma_f16(aF[i], bF[j], acc[i][j]);
    __syncthreads();
  }

  // epilogue
#pragma unroll
  for (int i = 0; i < 2; ++i) {
#pragma unroll
    for (int j = 0; j < 2; ++j) {
      int col = n0 + wx * 32 + j * 16 + lm;
#pragma unroll
      for (int r = 0; r < 8; ++r) {
        int row = m0 + wy * 32 + i * 16 + r + 8 * hlf;
        float v = acc[i][j][r];
        if constexpr (EP == 0) {
          outH[(size_t)row * N + col] = (_Float16)v;
        } else if constexpr (EP == 1) {
          outF[(size_t)row * N + col] = v + bias[col] + res[(size_t)row * N + col];
        } else if constexpr (EP == 2) {
          float t = v + bias[col];
          float gl = 0.5f * t * (1.0f + erff(t * 0.70710678f));
          outH[(size_t)row * N + col] = (_Float16)gl;
        } else if constexpr (EP == 3) {
          outH[(size_t)row * N + col] =
              (_Float16)(v + bias[col] + res[(size_t)row * N + col]);
        } else {  // EP == 4: transposed conv-out + input residual
          int bb = row / NTOK;
          int p  = row - bb * NTOK;
          size_t idx = ((size_t)bb * CCH + col) * NTOK + p;
          outF[idx] = v + bias[col] + xin[idx];
        }
      }
    }
  }
}

// ---------------------------------------------------------------------------
// Flash attention: 8 waves/block; wave w owns i-tile (blockIdx.x*8 + w).
// K/V 32-row tiles are async-staged into LDS once per block and feed all 8
// waves (8x less global K/V traffic, 64 WMMAs per staged tile).
// Per 32-col j-block per wave: 4 WMMAs for S=Q*K^T, online softmax with
// 16-lane shuffle reductions, P repacked via LDS to A-fragment layout,
// 4 WMMAs for O += P*V.
// ---------------------------------------------------------------------------
__global__ __launch_bounds__(256) void flash_attn_kernel(
    const _Float16* __restrict__ Q, const _Float16* __restrict__ Km,
    const _Float16* __restrict__ V, _Float16* __restrict__ O) {
  const int tid  = threadIdx.x;
  const int lane = tid & 31;
  const int wid  = tid >> 5;                 // 0..7
  const int h    = blockIdx.y;
  const int bb   = blockIdx.z;
  const int hlf  = lane >> 4;
  const int lm   = lane & 15;
  const int i0   = (blockIdx.x * 8 + wid) * 16;   // wave's 16-row i-tile
  const size_t baseRow = (size_t)bb * NTOK;
  const int hoff = h * DHEAD;
  const float scale = 0.125f;  // DIM_HEAD^-0.5

  __shared__ _Float16 Ks[32][72];      // 32 x 64 (+8 pad), row 144B
  __shared__ _Float16 Vs[32][72];
  __shared__ _Float16 Ps[8][16][40];   // per-wave 16x32 P tile

  // staging coords: 256 threads x 16B = one 32x64 f16 tile
  const int sr  = tid >> 3;        // 0..31
  const int scg = (tid & 7) * 8;   // 0..56 halfs
  const uint32_t kDst = lds_addr(&Ks[sr][scg]);
  const uint32_t vDst = lds_addr(&Vs[sr][scg]);

  // Q A-fragments (two 32-wide d chunks)
  v16h aQ[2];
  const _Float16* qrow = Q + (baseRow + i0 + lm) * INNER + hoff;
#pragma unroll
  for (int c = 0; c < 2; ++c) {
#pragma unroll
    for (int v = 0; v < 8; ++v) {
      int kk = (v < 4) ? (hlf * 8 + 2 * v) : (16 + hlf * 8 + 2 * (v - 4));
      v2h pr = *(const v2h*)(qrow + c * 32 + kk);
      aQ[c][2 * v]     = pr[0];
      aQ[c][2 * v + 1] = pr[1];
    }
  }

  float mrow[8], lrow[8];
  v8f o[4];
#pragma unroll
  for (int r = 0; r < 8; ++r) { mrow[r] = -1e30f; lrow[r] = 0.0f; }
  o[0] = {}; o[1] = {}; o[2] = {}; o[3] = {};

  for (int jb = 0; jb < NTOK; jb += 32) {
    // async-stage K and V 32-row tiles
    const size_t srow = (baseRow + jb + sr) * INNER + hoff + scg;
    async_copy_b128(kDst, Km + srow);
    async_copy_b128(vDst, V + srow);
    wait_async0();
    __syncthreads();

    // S = Q * K^T for this 32-column block (two 16-col WMMA tiles)
    v8f s[2];
#pragma unroll
    for (int u = 0; u < 2; ++u) {
      v16h bK0, bK1;
      const int jj = u * 16 + lm;   // K row within tile (= output col j)
#pragma unroll
      for (int v = 0; v < 8; ++v) {
        int dd = (v < 4) ? (hlf * 8 + 2 * v) : (16 + hlf * 8 + 2 * (v - 4));
        v2h p0 = *(const v2h*)&Ks[jj][dd];
        v2h p1 = *(const v2h*)&Ks[jj][32 + dd];
        bK0[2 * v] = p0[0]; bK0[2 * v + 1] = p0[1];
        bK1[2 * v] = p1[0]; bK1[2 * v + 1] = p1[1];
      }
      v8f z = {};
      z = wmma_f16(aQ[0], bK0, z);
      z = wmma_f16(aQ[1], bK1, z);
      s[u] = z;
    }

    // online softmax update (lane: col j = u*16 + lm, rows r + 8*hlf)
    float p0[8], p1[8];
#pragma unroll
    for (int r = 0; r < 8; ++r) {
      float sv0 = s[0][r] * scale;
      float sv1 = s[1][r] * scale;
      float mx = fmaxf(sv0, sv1);
      for (int d = 1; d < 16; d <<= 1) mx = fmaxf(mx, __shfl_xor(mx, d, 32));
      float mnew = fmaxf(mrow[r], mx);
      p0[r] = __expf(sv0 - mnew);
      p1[r] = __expf(sv1 - mnew);
      float sum = p0[r] + p1[r];
      for (int d = 1; d < 16; d <<= 1) sum += __shfl_xor(sum, d, 32);
      float cf = __expf(mrow[r] - mnew);
      lrow[r] = lrow[r] * cf + sum;
      mrow[r] = mnew;
#pragma unroll
      for (int c = 0; c < 4; ++c) o[c][r] *= cf;
    }

    // repack P (C-fragment layout) -> A-fragment layout via per-wave LDS
#pragma unroll
    for (int r = 0; r < 8; ++r) {
      int prow = r + 8 * hlf;
      Ps[wid][prow][lm]      = (_Float16)p0[r];
      Ps[wid][prow][16 + lm] = (_Float16)p1[r];
    }
    __syncthreads();
    v16h aP;
#pragma unroll
    for (int v = 0; v < 8; ++v) {
      int kk = (v < 4) ? (hlf * 8 + 2 * v) : (16 + hlf * 8 + 2 * (v - 4));
      v2h pr = *(const v2h*)&Ps[wid][lm][kk];
      aP[2 * v]     = pr[0];
      aP[2 * v + 1] = pr[1];
    }

    // O += P * V  (four 16-col d chunks) from LDS V tile
#pragma unroll
    for (int c = 0; c < 4; ++c) {
      v16h bV;
#pragma unroll
      for (int v = 0; v < 8; ++v) {
        int kd = (v < 4) ? (hlf * 8 + 2 * v) : (16 + hlf * 8 + 2 * (v - 4));
        bV[2 * v]     = Vs[kd][c * 16 + lm];
        bV[2 * v + 1] = Vs[kd + 1][c * 16 + lm];
      }
      o[c] = wmma_f16(aP, bV, o[c]);
    }
    __syncthreads();   // protect Ks/Vs/Ps before next iteration's staging
  }

  // normalize and store f16
#pragma unroll
  for (int c = 0; c < 4; ++c) {
#pragma unroll
    for (int r = 0; r < 8; ++r) {
      int row = i0 + r + 8 * hlf;
      float val = o[c][r] / lrow[r];
      O[(baseRow + row) * INNER + hoff + c * 16 + lm] = (_Float16)val;
    }
  }
}

// ---------------------------------------------------------------------------
// Host launcher
// ---------------------------------------------------------------------------
extern "C" void kernel_launch(void* const* d_in, const int* in_sizes, int n_in,
                              void* d_out, int out_size, void* d_ws,
                              size_t ws_size, hipStream_t stream) {
  const float* x   = (const float*)d_in[0];
  const float* Wq  = (const float*)d_in[1];
  const float* Wk  = (const float*)d_in[2];
  const float* Wv  = (const float*)d_in[3];
  const float* Wo  = (const float*)d_in[4];
  const float* bo  = (const float*)d_in[5];
  const float* g1  = (const float*)d_in[6];
  const float* b1  = (const float*)d_in[7];
  const float* g2  = (const float*)d_in[8];
  const float* b2  = (const float*)d_in[9];
  const float* Wf1 = (const float*)d_in[10];
  const float* bf1 = (const float*)d_in[11];
  const float* Wf2 = (const float*)d_in[12];
  const float* bf2 = (const float*)d_in[13];
  const float* Wp  = (const float*)d_in[14];
  const float* bp  = (const float*)d_in[15];

  uintptr_t p = (uintptr_t)d_ws;
  auto carve = [&](size_t bytes) -> void* {
    void* r = (void*)p;
    p += (bytes + 255) & ~(size_t)255;
    return r;
  };
  float*    T    = (float*)carve((size_t)BN * CCH * 4);
  float*    T2   = (float*)carve((size_t)BN * CCH * 4);
  _Float16* TN   = (_Float16*)carve((size_t)BN * CCH * 2);
  _Float16* Qh   = (_Float16*)carve((size_t)BN * INNER * 2);
  _Float16* Kh   = (_Float16*)carve((size_t)BN * INNER * 2);
  _Float16* Vh   = (_Float16*)carve((size_t)BN * INNER * 2);
  _Float16* Oh   = (_Float16*)carve((size_t)BN * INNER * 2);
  _Float16* Fh   = (_Float16*)carve((size_t)BN * INNER * 2);
  _Float16* T3h  = (_Float16*)carve((size_t)BN * CCH * 2);
  _Float16* WqH  = (_Float16*)carve((size_t)CCH * INNER * 2);
  _Float16* WkH  = (_Float16*)carve((size_t)CCH * INNER * 2);
  _Float16* WvH  = (_Float16*)carve((size_t)CCH * INNER * 2);
  _Float16* WoH  = (_Float16*)carve((size_t)INNER * CCH * 2);
  _Float16* Wf1H = (_Float16*)carve((size_t)CCH * INNER * 2);
  _Float16* Wf2H = (_Float16*)carve((size_t)INNER * CCH * 2);
  _Float16* WpH  = (_Float16*)carve((size_t)CCH * CCH * 2);

  auto cvt = [&](const float* s, _Float16* d, int n) {
    f32_to_f16_kernel<<<(n + 255) / 256, 256, 0, stream>>>(s, d, n);
  };
  cvt(Wq, WqH, CCH * INNER);
  cvt(Wk, WkH, CCH * INNER);
  cvt(Wv, WvH, CCH * INNER);
  cvt(Wo, WoH, INNER * CCH);
  cvt(Wf1, Wf1H, CCH * INNER);
  cvt(Wf2, Wf2H, INNER * CCH);
  cvt(Wp, WpH, CCH * CCH);

  // 1) transpose + LN1
  transpose_ln_kernel<<<BN, 256, 0, stream>>>(x, g1, b1, T, TN);

  // 2) Q/K/V projections  [9216x256] x [256x512]
  dim3 blk(256);
  dim3 gQKV(INNER / 128, BN / 64);
  gemm_kernel<0><<<gQKV, blk, 0, stream>>>(TN, WqH, nullptr, nullptr, nullptr,
                                           Qh, nullptr, BN, INNER, CCH);
  gemm_kernel<0><<<gQKV, blk, 0, stream>>>(TN, WkH, nullptr, nullptr, nullptr,
                                           Kh, nullptr, BN, INNER, CCH);
  gemm_kernel<0><<<gQKV, blk, 0, stream>>>(TN, WvH, nullptr, nullptr, nullptr,
                                           Vh, nullptr, BN, INNER, CCH);

  // 3) flash attention: 8 i-tiles per block
  dim3 gAttn(NTOK / 16 / 8, NHEAD, BATCH);
  flash_attn_kernel<<<gAttn, dim3(256), 0, stream>>>(Qh, Kh, Vh, Oh);

  // 4) out projection + bias + residual -> T2 (f32)
  dim3 gOut(CCH / 128, BN / 64);
  gemm_kernel<1><<<gOut, blk, 0, stream>>>(Oh, WoH, bo, T, T2, nullptr, nullptr,
                                           BN, CCH, INNER);

  // 5) LN2
  ln_kernel<<<BN, 256, 0, stream>>>(T2, g2, b2, TN);

  // 6) FFN1 + GELU -> Fh (f16)
  gemm_kernel<2><<<gQKV, blk, 0, stream>>>(TN, Wf1H, bf1, nullptr, nullptr, Fh,
                                           nullptr, BN, INNER, CCH);

  // 7) FFN2 + bias + residual(T2) -> T3h (f16)
  gemm_kernel<3><<<gOut, blk, 0, stream>>>(Fh, Wf2H, bf2, T2, nullptr, T3h,
                                           nullptr, BN, CCH, INNER);

  // 8) 1x1 conv + bp + input residual, transposed store -> d_out (f32)
  gemm_kernel<4><<<gOut, blk, 0, stream>>>(T3h, WpH, bp, nullptr, (float*)d_out,
                                           nullptr, x, BN, CCH, CCH);
}
